// Simple4BitLinear_7112465842827
// MI455X (gfx1250) — matrix operand order repaired and checked by reading, not compiled, not run on hardware
//
#include <hip/hip_runtime.h>

typedef __attribute__((ext_vector_type(16))) _Float16 v16h;
typedef __attribute__((ext_vector_type(8)))  _Float16 v8h;
typedef __attribute__((ext_vector_type(4)))  _Float16 v4h;
typedef __attribute__((ext_vector_type(8)))  float    v8f;

#define M_DIM 8192      // B*S = 4*2048
#define N_DIM 16384     // OUT
#define K_DIM 4096      // IN
#define BM 128
#define BN 128
#define BK 64
#define NK (K_DIM / BK) // 64
#define LDSS 72         // 64 + 8 halves pad (144B row stride)
#define BUFH (BM * LDSS)   // halves per buffer (same for x and w: 9216)

// y[m][o] = scale[o]*sum_k x[m][k]*nib[o][k] + rowsum_x[m]*zp[o] + bias[o]
__global__ __launch_bounds__(256)
void w4a16_wmma_gemm(const float* __restrict__ x,
                     const int*   __restrict__ packed,
                     const float* __restrict__ scale,
                     const float* __restrict__ zp,
                     const float* __restrict__ bias,
                     float* __restrict__ out)
{
    __shared__ __align__(16) _Float16 xs[2 * BUFH];   // double buffered
    __shared__ __align__(16) _Float16 wsh[2 * BUFH];  // double buffered
    __shared__ float rparts[BM * 16];                 // [row][16] fp32 partials
    __shared__ float rsum[BM];

    const int tid   = threadIdx.x;
    const int wave  = tid >> 5;        // 0..7
    const int lane  = tid & 31;
    const int lhalf = lane >> 4;       // 0/1
    const int lcol  = lane & 15;
    const int waveM = wave >> 2;       // 0..1 (64 rows)
    const int waveN = wave & 3;        // 0..3 (32 cols)

    const int n0 = blockIdx.x * BN;
    const int m0 = blockIdx.y * BM;

    // ---- per-thread fill geometry (chunk strides are compile-time constants) ----
    const int xrow0 = tid >> 4;                 // 0..15, chunk stride +16 rows
    const int xc16  = tid & 15;                 // float4 column slot
    const int xc4   = xc16 << 2;                // float col 0,4,..,60
    const float* px0 = x + (size_t)(m0 + xrow0) * K_DIM + xc4;
    _Float16* xs0 = xs + xrow0 * LDSS + xc4;

    const int wrow0 = tid >> 3;                 // 0..31, chunk stride +32 rows
    const int wj4   = (tid & 7) << 2;           // int32 col 0,4,..,28
    const int* pw0 = packed + (size_t)(n0 + wrow0) * (K_DIM / 2) + wj4;
    _Float16* ws0 = wsh + wrow0 * LDSS + (wj4 << 1);

    // per-wave fragment read bases (offset by rdo each iteration)
    const _Float16* xfrag0 = xs + (waveM * 64 + lcol) * LDSS + lhalf * 8;
    const _Float16* wfrag0 = wsh + (waveN * 32 + lcol) * LDSS + lhalf * 8;

    v8f acc[4][2];
    {
        v8f z = {};
        #pragma unroll
        for (int i = 0; i < 4; ++i)
            #pragma unroll
            for (int j = 0; j < 2; ++j)
                acc[i][j] = z;
    }

    // fp32 rowsum partials: rsch[i] covers row xrow0+16*i, cols [xc4, xc4+4), all k
    float rsch[8];
    #pragma unroll
    for (int i = 0; i < 8; ++i) rsch[i] = 0.0f;

    float4 xv[8];
    int4   wv[4];

    // staged-register global fetch of one 64-deep tile
    auto fetch = [&]() {
        #pragma unroll
        for (int i = 0; i < 8; ++i)
            xv[i] = *(const float4*)(px0 + (size_t)i * 16 * K_DIM);
        #pragma unroll
        for (int i = 0; i < 4; ++i)
            wv[i] = *(const int4*)(pw0 + (size_t)i * 32 * (K_DIM / 2));
        px0 += BK;
        pw0 += BK / 2;
    };
    // drain staged registers into LDS at half-offset `off` (f16 convert) and
    // fold this tile's fp32 values into the register-resident rowsum partials.
    auto drain = [&](int off) {
        _Float16* xd = xs0 + off;
        _Float16* wd = ws0 + off;
        #pragma unroll
        for (int i = 0; i < 8; ++i) {
            v4h h;
            h[0] = (_Float16)xv[i].x; h[1] = (_Float16)xv[i].y;
            h[2] = (_Float16)xv[i].z; h[3] = (_Float16)xv[i].w;
            *(v4h*)(xd + i * 16 * LDSS) = h;
            rsch[i] += xv[i].x; rsch[i] += xv[i].y;   // 8 independent 4-add chains
            rsch[i] += xv[i].z; rsch[i] += xv[i].w;
        }
        #pragma unroll
        for (int i = 0; i < 4; ++i) {
            v8h h;
            int b;
            b = wv[i].x; h[0] = (_Float16)(b & 15); h[1] = (_Float16)((b >> 4) & 15);
            b = wv[i].y; h[2] = (_Float16)(b & 15); h[3] = (_Float16)((b >> 4) & 15);
            b = wv[i].z; h[4] = (_Float16)(b & 15); h[5] = (_Float16)((b >> 4) & 15);
            b = wv[i].w; h[6] = (_Float16)(b & 15); h[7] = (_Float16)((b >> 4) & 15);
            *(v8h*)(wd + i * 32 * LDSS) = h;
        }
    };

    // ---- pipeline prologue: tile0 -> buf0, tile1 in flight ----
    fetch();
    drain(0);
    fetch();            // tile 1, consumed at top of iteration kb=0
    __syncthreads();    // buf0 ready

    int rdo = 0, wro = BUFH;   // uniform read/write buffer offsets (halves)

    for (int kb = 0; kb < NK; ++kb) {
        // ---- overlap: drain tile kb+1 into the other buffer, then issue kb+2 loads
        if (kb + 1 < NK) {
            drain(wro);                     // waits on tile kb+1 global loads
            if (kb + 2 < NK) fetch();       // tile kb+2 flies during WMMA below
        }

        // ---- WMMA: two 16x16x32 k-steps over the 64-deep tile ----
        #pragma unroll
        for (int ks = 0; ks < 2; ++ks) {
            // A(16x32 f16): row = l&15; halves k = ks*32 + 8*(l>=16) + {0..7,16..23}
            v16h afrag[4];
            #pragma unroll
            for (int tm = 0; tm < 4; ++tm) {
                const _Float16* base = xfrag0 + rdo + tm * 16 * LDSS + ks * 32;
                v8h lo = *(const v8h*)(base);
                v8h hi = *(const v8h*)(base + 16);
                #pragma unroll
                for (int e = 0; e < 8; ++e) { afrag[tm][e] = lo[e]; afrag[tm][e + 8] = hi[e]; }
            }
            // B(32x16 f16): col = l&15; k = ks*32 + 8*(l>=16) + {0..7,16..23}
            v16h bfrag[2];
            #pragma unroll
            for (int tn = 0; tn < 2; ++tn) {
                const _Float16* base = wfrag0 + rdo + tn * 16 * LDSS + ks * 32;
                v8h lo = *(const v8h*)(base);
                v8h hi = *(const v8h*)(base + 16);
                #pragma unroll
                for (int e = 0; e < 8; ++e) { bfrag[tn][e] = lo[e]; bfrag[tn][e + 8] = hi[e]; }
            }
            #pragma unroll
            for (int tm = 0; tm < 4; ++tm)
                #pragma unroll
                for (int tn = 0; tn < 2; ++tn)
                    acc[tm][tn] = __builtin_amdgcn_wmma_f32_16x16x32_f16(
                        false, afrag[tm], false, bfrag[tn],
                        (short)0, acc[tm][tn], false, false);
        }

        // one barrier per iteration: publishes buf[wro], retires reads of buf[rdo]
        __syncthreads();
        const int t = rdo; rdo = wro; wro = t;   // uniform swap
    }

    // ---- rowsum reduction (once, in the epilogue) ----
    #pragma unroll
    for (int i = 0; i < 8; ++i)
        rparts[(xrow0 + 16 * i) * 16 + xc16] = rsch[i];
    __syncthreads();
    if (tid < BM) {
        float t = 0.0f;
        #pragma unroll
        for (int c = 0; c < 16; ++c) t += rparts[tid * 16 + c];
        rsum[tid] = t;
    }
    __syncthreads();

    // ---- writeout: y = sc*acc + rowsum*zp + bias (NT stores keep L2 for operands) ----
    #pragma unroll
    for (int tn = 0; tn < 2; ++tn) {
        const int gcol = n0 + waveN * 32 + tn * 16 + lcol;
        const float sc = scale[gcol];
        const float z  = zp[gcol];
        const float bi = bias[gcol];
        #pragma unroll
        for (int tm = 0; tm < 4; ++tm) {
            #pragma unroll
            for (int r = 0; r < 8; ++r) {
                const int mloc = waveM * 64 + tm * 16 + lhalf * 8 + r;  // C layout: M = r + 8*(lane>=16)
                const float v = sc * acc[tm][tn][r] + rsum[mloc] * z + bi;
                __builtin_nontemporal_store(v, &out[(size_t)(m0 + mloc) * N_DIM + gcol]);
            }
        }
    }
}

extern "C" void kernel_launch(void* const* d_in, const int* in_sizes, int n_in,
                              void* d_out, int out_size, void* d_ws, size_t ws_size,
                              hipStream_t stream) {
    (void)in_sizes; (void)n_in; (void)out_size; (void)d_ws; (void)ws_size;
    const float* x      = (const float*)d_in[0];
    const int*   packed = (const int*)  d_in[1];
    const float* scale  = (const float*)d_in[2];
    const float* zpnt   = (const float*)d_in[3];
    const float* bias   = (const float*)d_in[4];
    float* out = (float*)d_out;

    dim3 grid(N_DIM / BN, M_DIM / BM);   // 128 x 64 workgroups
    dim3 block(256);                      // 8 waves (wave32)
    w4a16_wmma_gemm<<<grid, block, 0, stream>>>(x, packed, scale, zpnt, bias, out);
}
